// PointConv2d_35648228557117
// MI455X (gfx1250) — matrix operand order repaired and checked by reading, not compiled, not run on hardware
//
#include <hip/hip_runtime.h>

// Problem constants (from reference setup_inputs)
#define B_    8
#define N_    16384
#define KNN   9
#define C_    64
#define CO_   64
#define CIN   66          // C + 2
#define FAN   594         // 9 * 66
#define KPAD  608         // 19 * 32  (K padded for bf16 WMMA chunks)
#define NKCH  19          // KPAD / 32
#define ASTR  616         // A-buffer k-stride in ushorts (pad; 1232B = 77*16, keeps b128 align)
#define MT    2           // M-tiles (of 16 points) per wave

typedef __attribute__((ext_vector_type(16))) __bf16 v16bf;
typedef __attribute__((ext_vector_type(8)))  float  v8f;

__device__ __forceinline__ unsigned short f2bf(float f) {
  unsigned int u = __float_as_uint(f);
  u += 0x7FFFu + ((u >> 16) & 1u);          // round-to-nearest-even
  return (unsigned short)(u >> 16);
}
__device__ __forceinline__ float leaky(float x) { return x >= 0.f ? x : 0.1f * x; }

// ---------------------------------------------------------------------------
// Kernel 0: pack lin_w (fp32 [64][594]) into bf16 WMMA B-fragment layout.
// B 32x16 bf16 layout used here (contiguous k, same as A): lane L: N = L%16;
// lanes 0-15 hold K=0..15 (VGPR v packs K=2v,2v+1), lanes 16-31 hold K=16..31.
// Fragment element (kk, nt, lane, v, hi) -> B[k][n] = lin_w[n*594 + k],
//   n = nt*16 + lane%16 ; k = kk*32 + 16*(lane/16) + 2v + hi ; zero-pad k>=594.
// ---------------------------------------------------------------------------
__global__ void pack_linB(const float* __restrict__ lin_w,
                          unsigned short* __restrict__ pB) {
  int t = blockIdx.x * blockDim.x + threadIdx.x;
  if (t >= NKCH * 4 * 32) return;
  int lane = t & 31;
  int nt   = (t >> 5) & 3;
  int kk   = t >> 7;
  int n    = nt * 16 + (lane & 15);
  int half = lane >> 4;
  unsigned short vals[16];
#pragma unroll
  for (int v = 0; v < 8; ++v) {
#pragma unroll
    for (int hi = 0; hi < 2; ++hi) {
      int k = kk * 32 + half * 16 + 2 * v + hi;
      float f = (k < FAN) ? lin_w[n * FAN + k] : 0.f;
      vals[2 * v + hi] = f2bf(f);
    }
  }
  uint4* dst = (uint4*)(pB + (size_t)t * 16);
  dst[0] = *(uint4*)&vals[0];
  dst[1] = *(uint4*)&vals[8];
}

// ---------------------------------------------------------------------------
// Main kernel: one wave (32 threads) = 32 consecutive points (2 M-tiles).
// Stage A: WeightNet (VALU fp32)  -> s_w[m][o][k]
// Stage B: wf = w @ knn (VALU fp32), bf16 -> s_a[tile][m][k=o*66+c]
// Stage C: per N-tile: single in-place acc chain over 19 K-chunks,
//          each B fragment (global b128 x2) feeds 2 WMMAs (one per M-tile).
// Output : bias + leaky into LDS transpose buffer, coalesced b128 stores.
// ---------------------------------------------------------------------------
__global__ __launch_bounds__(32)
void pointconv_main(const float* __restrict__ xy,
                    const float* __restrict__ feat,
                    const float* __restrict__ w1,
                    const float* __restrict__ b1,
                    const float* __restrict__ w2,
                    const float* __restrict__ b2,
                    const float* __restrict__ lin_b,
                    const unsigned short* __restrict__ pB,
                    float* __restrict__ out) {
  __shared__ float          s_w[MT * 16 * 81];        // [m][o][k] fp32 (10368 B)
  __shared__ unsigned short s_a[MT * 16 * ASTR];      // bf16 A tiles  (39424 B)

  const int lane = threadIdx.x;
  const int p0   = blockIdx.x * (MT * 16);            // first flat point
  const int b    = p0 / N_;
  const int n0   = p0 % N_;                           // never crosses b boundary
  const float* xyb   = xy   + (size_t)b * 2  * N_ * KNN;
  const float* featb = feat + (size_t)b * C_ * N_ * KNN;

  // ---- Stage A: WeightNet, item = (m, k), MT*16*9 items ----
  for (int item = lane; item < MT * 16 * KNN; item += 32) {
    int m = item / KNN, k = item % KNN;
    int base = (n0 + m) * KNN + k;
    float x = xyb[base];
    float y = xyb[N_ * KNN + base];
    float h[9];
#pragma unroll
    for (int o = 0; o < 9; ++o)
      h[o] = leaky(w1[o * 2 + 0] * x + w1[o * 2 + 1] * y + b1[o]);
#pragma unroll
    for (int o = 0; o < 9; ++o) {
      float acc = b2[o];
#pragma unroll
      for (int j = 0; j < 9; ++j) acc += w2[o * 9 + j] * h[j];
      s_w[(m * 9 + o) * 9 + k] = leaky(acc);
    }
  }

  // zero the K pad region (594..607) of both A tiles
  for (int i = lane; i < MT * 16 * (KPAD - FAN); i += 32) {
    int m = i / (KPAD - FAN), k = FAN + i % (KPAD - FAN);
    s_a[(m / 16) * (16 * ASTR) + (m & 15) * ASTR + k] = 0;
  }
  __syncthreads();

  // ---- Stage B: wf[m][o][c] = sum_k w[m][o][k]*knn[m][k][c]; item=(m,c) ----
  for (int item = lane; item < MT * 16 * CIN; item += 32) {
    int m = item / CIN, c = item % CIN;
    int base = (n0 + m) * KNN;
    const float* src = (c < 2) ? (xyb + (size_t)c * N_ * KNN + base)
                               : (featb + (size_t)(c - 2) * N_ * KNN + base);
    float f[9];
#pragma unroll
    for (int k = 0; k < 9; ++k) f[k] = src[k];
#pragma unroll
    for (int o = 0; o < 9; ++o) {
      float acc = 0.f;
#pragma unroll
      for (int k = 0; k < 9; ++k) acc += s_w[(m * 9 + o) * 9 + k] * f[k];
      s_a[(m / 16) * (16 * ASTR) + (m & 15) * ASTR + o * CIN + c] = f2bf(acc);
    }
  }
  __syncthreads();

  // ---- Stage C: bf16 WMMA, nt outer so each acc chain stays in place ----
  float* s_o = s_w;                                   // reuse: [MT][64][16] f32
  const int half = lane >> 4;
  const int mrow = lane & 15;
  const v8f vzero = {0.f,0.f,0.f,0.f,0.f,0.f,0.f,0.f};

  for (int nt = 0; nt < 4; ++nt) {
    v8f acc0 = vzero;
    v8f acc1 = vzero;
    // pB layout: [kk][nt][lane][16 ushorts]; uint4 index, stride 256 per kk
    const uint4* bsrc = (const uint4*)pB + ((size_t)nt * 32 + lane) * 2;
    for (int kk = 0; kk < NKCH; ++kk) {
      union { v16bf v; uint4 q[2]; } bf;
      bf.q[0] = bsrc[(size_t)kk * 256 + 0];
      bf.q[1] = bsrc[(size_t)kk * 256 + 1];
      int aoff = mrow * ASTR + kk * 32 + 16 * half;   // byte-aligned to 16
      union { v16bf v; uint4 q[2]; } a0, a1;
      a0.q[0] = *(const uint4*)&s_a[aoff];
      a0.q[1] = *(const uint4*)&s_a[aoff + 8];
      a1.q[0] = *(const uint4*)&s_a[16 * ASTR + aoff];
      a1.q[1] = *(const uint4*)&s_a[16 * ASTR + aoff + 8];
      acc0 = __builtin_amdgcn_wmma_f32_16x16x32_bf16(
          false, a0.v, false, bf.v, (short)0, acc0, false, false);
      acc1 = __builtin_amdgcn_wmma_f32_16x16x32_bf16(
          false, a1.v, false, bf.v, (short)0, acc1, false, false);
    }
    // drain this nt: bias + leaky -> LDS transpose buffer (frees the accs)
    int co = nt * 16 + mrow;
    float bias = lin_b[co];
#pragma unroll
    for (int v = 0; v < 8; ++v) {
      int m = v + 8 * half;                           // C/D layout: M = v + 8*(L/16)
      s_o[co * 16 + m]        = leaky(acc0[v] + bias);
      s_o[1024 + co * 16 + m] = leaky(acc1[v] + bias);
    }
  }
  __syncthreads();

  // ---- coalesced stores: lane owns rows co = lane, lane+32 per tile ----
#pragma unroll
  for (int t = 0; t < MT; ++t) {
#pragma unroll
    for (int r = 0; r < 2; ++r) {
      int co = r * 32 + lane;
      float4* dst = (float4*)(out + (size_t)b * CO_ * N_ + (size_t)co * N_ + n0 + t * 16);
      const float4* srcv = (const float4*)(s_o + t * 1024 + co * 16);
      dst[0] = srcv[0]; dst[1] = srcv[1]; dst[2] = srcv[2]; dst[3] = srcv[3];
    }
  }
}

// ---------------------------------------------------------------------------
extern "C" void kernel_launch(void* const* d_in, const int* in_sizes, int n_in,
                              void* d_out, int out_size, void* d_ws, size_t ws_size,
                              hipStream_t stream) {
  const float* xy   = (const float*)d_in[0];   // [8,2,16384,9]
  const float* feat = (const float*)d_in[1];   // [8,64,16384,9]
  const float* w1   = (const float*)d_in[2];   // [9,2]
  const float* b1   = (const float*)d_in[3];   // [9]
  const float* w2   = (const float*)d_in[4];   // [9,9]
  const float* b2   = (const float*)d_in[5];   // [9]
  const float* lw   = (const float*)d_in[6];   // [64,594]
  const float* lb   = (const float*)d_in[7];   // [64]
  float* out = (float*)d_out;                  // [8,64,16384]
  unsigned short* pB = (unsigned short*)d_ws;  // needs 608*64*2 = 77824 B

  (void)in_sizes; (void)n_in; (void)out_size; (void)ws_size;

  int packThreads = NKCH * 4 * 32;             // 2432
  pack_linB<<<(packThreads + 127) / 128, 128, 0, stream>>>(lw, pB);

  int tiles = (B_ * N_) / (MT * 16);           // 4096 waves, 1 wave per block
  pointconv_main<<<tiles, 32, 0, stream>>>(xy, feat, w1, b1, w2, b2, lb, pB, out);
}
